// VectorQuantiser_77369540870566
// MI455X (gfx1250) — matrix operand order repaired and checked by reading, not compiled, not run on hardware
//
#include <hip/hip_runtime.h>
#include <hip/hip_bf16.h>

// ---------------------------------------------------------------------------
// VQ-VAE vector quantiser forward for gfx1250 (MI455X, wave32, WMMA).
// Heavy op: S = flat[16384,1024] @ E[1024,1024] via split-bf16 WMMA
// (hi*hi + hi*lo + lo*hi) -> near-fp32 distances for exact-ish argmin.
// Register blocking: each wave computes 64 rows x 64 cols per pass
// (4x4 fragments, 48 WMMAs per 32-deep D-chunk -> 16 FLOP per operand byte).
// ---------------------------------------------------------------------------

typedef __attribute__((ext_vector_type(16))) __bf16 v16bf;
typedef __attribute__((ext_vector_type(8)))  float  v8f;

#define BB   64
#define HH   32
#define WW   32
#define CC   256
#define DD   1024          // H*W
#define KK   1024          // codebook size
#define MM   16384         // B*C
#define NELEM 16777216     // B*H*W*C

static __device__ __forceinline__ unsigned short f32_to_bf16(float f) {
    unsigned int u = __float_as_uint(f);
    unsigned int r = (u + 0x7fffu + ((u >> 16) & 1u)) >> 16;   // RNE
    return (unsigned short)r;
}
static __device__ __forceinline__ float bf16_to_f32(unsigned short h) {
    return __uint_as_float(((unsigned int)h) << 16);
}

// ---------------------------------------------------------------------------
// Kernel 1: column norms of E + zero the loss accumulator.
// ---------------------------------------------------------------------------
__global__ void vq_prep_e2(const float* __restrict__ E,
                           float* __restrict__ e2,
                           float* __restrict__ lossp) {
    int n = blockIdx.x * 256 + threadIdx.x;   // 0..1023
    float s = 0.f;
    for (int d = 0; d < DD; ++d) {
        float v = E[(size_t)d * KK + n];
        s += v * v;
    }
    e2[n] = s;
    if (n == 0) *lossp = 0.f;
}

// ---------------------------------------------------------------------------
// Kernel 2: E (f32 [D,K] row-major) -> bf16 hi/lo packed in WMMA B-fragment
// layout: EB[nt][dc][lane][16].  16-bit B 32x16: lanes 0-15 hold K={0..7,16..23},
// lanes 16-31 hold K={8..15,24..31}; column n = lane%16.
// ---------------------------------------------------------------------------
__global__ void vq_prep_e(const float* __restrict__ E,
                          unsigned short* __restrict__ ebhi,
                          unsigned short* __restrict__ eblo) {
    __shared__ float tile[32][257];
    int blk = blockIdx.x;            // 0..127
    int dc  = blk >> 2;              // 0..31
    int nb  = blk & 3;               // 0..3 (256-wide n blocks)
    int t   = threadIdx.x;           // 0..255
    int d0  = dc * 32, n0 = nb * 256;

    for (int k = 0; k < 32; ++k)
        tile[k][t] = E[(size_t)(d0 + k) * KK + n0 + t];   // coalesced
    __syncthreads();

    for (int s = t; s < 512; s += 256) {
        int ntl = s >> 5;            // 0..15 local n-tile
        int L   = s & 31;            // lane slot
        int nt  = (n0 >> 4) + ntl;
        int col = ntl * 16 + (L & 15);
        size_t base = ((size_t)nt * 32 + dc) * 512 + (size_t)L * 16;
        for (int j = 0; j < 16; ++j) {
            int dOff = (L < 16) ? ((j < 8) ? j : j + 8)
                                : ((j < 8) ? j + 8 : j + 16);
            float v = tile[dOff][col];
            unsigned short h = f32_to_bf16(v);
            unsigned short l = f32_to_bf16(v - bf16_to_f32(h));
            ebhi[base + j] = h;
            eblo[base + j] = l;
        }
    }
}

// ---------------------------------------------------------------------------
// Kernel 3: x [B,H,W,C] f32 -> bf16 hi/lo packed in WMMA A-fragment layout:
// XA[mtile][dc][lane][16], m = b*C + c, d = h*W + w.
// 16-bit A 16x32: row M = lane%16; lane<16: K = {0..7,16..23};
// lane>=16: K = {8..15,24..31}.
// ---------------------------------------------------------------------------
__global__ void vq_prep_x(const float* __restrict__ x,
                          unsigned short* __restrict__ xahi,
                          unsigned short* __restrict__ xalo) {
    __shared__ float tile[32][257];
    int blk = blockIdx.x;            // 0..2047
    int b   = blk >> 5;              // 0..63
    int dc  = blk & 31;              // 0..31
    int t   = threadIdx.x;           // 0..255
    const float* src = x + ((size_t)b * DD + dc * 32) * CC;

    for (int k = 0; k < 32; ++k)
        tile[k][t] = src[(size_t)k * CC + t];             // coalesced over c
    __syncthreads();

    for (int s = t; s < 512; s += 256) {
        int mtl = s >> 5;            // 0..15 local m-tile (c group)
        int L   = s & 31;            // lane slot
        int c   = mtl * 16 + (L & 15);
        int mtile = b * 16 + mtl;
        size_t base = ((size_t)mtile * 32 + dc) * 512 + (size_t)L * 16;
        for (int j = 0; j < 16; ++j) {
            int dOff = (L < 16) ? ((j < 8) ? j : j + 8)
                                : ((j < 8) ? j + 8 : j + 16);
            float v = tile[dOff][c];
            unsigned short h = f32_to_bf16(v);
            unsigned short l = f32_to_bf16(v - bf16_to_f32(h));
            xahi[base + j] = h;
            xalo[base + j] = l;
        }
    }
}

// ---------------------------------------------------------------------------
// Kernel 4: fused GEMM (split-bf16 WMMA) + argmin, 4x4 register blocking.
// Grid: 128 blocks x 256 threads (8 waves = 2 m-groups x 4 N-quarters).
// Wave owns 4 m-frags (64 rows) and scans a 256-col quarter of the codebook
// in 4 passes of 64 cols (4 accumfrag columns). Per 32-deep D-chunk:
// 8 A-pair + 8 B-pair fragment loads feed 48 v_wmma_f32_16x16x32_bf16.
// Per-quarter argmin partials are merged across waves through LDS.
// C/D layout: VGPR r, lane L -> row = r + (L<16?0:8), col = n0 + L%16.
// ---------------------------------------------------------------------------
__global__ void __launch_bounds__(256)
vq_gemm_argmin(const unsigned short* __restrict__ xahi,
               const unsigned short* __restrict__ xalo,
               const unsigned short* __restrict__ ebhi,
               const unsigned short* __restrict__ eblo,
               const float* __restrict__ e2,
               int* __restrict__ idxout) {
    __shared__ float s_val[2][4][64];
    __shared__ int   s_idx[2][4][64];

    int wave   = threadIdx.x >> 5;
    int lane   = threadIdx.x & 31;
    int mgroup = wave & 1;           // 2 m-groups of 64 rows
    int nq     = wave >> 1;          // 4 codebook quarters of 256 cols
    int mt0    = blockIdx.x * 8 + mgroup * 4;   // first of 4 m-frags

    const v16bf* __restrict__ Ah0 = (const v16bf*)xahi + (size_t)mt0 * 1024 + lane;
    const v16bf* __restrict__ Al0 = (const v16bf*)xalo + (size_t)mt0 * 1024 + lane;
    const v16bf* __restrict__ Bh  = (const v16bf*)ebhi + lane;
    const v16bf* __restrict__ Bl  = (const v16bf*)eblo + lane;

    float minv[4][8];
    int   mini[4][8];
#pragma unroll
    for (int t = 0; t < 4; ++t)
#pragma unroll
        for (int r = 0; r < 8; ++r) { minv[t][r] = 3.0e38f; mini[t][r] = 0; }

    for (int p = 0; p < 4; ++p) {                 // 4 passes of 64 cols
        int nbk = nq * 4 + p;                     // 64-col block index 0..15
        v8f acc[4][4];                            // [m-frag][n-frag]
#pragma unroll
        for (int t = 0; t < 4; ++t)
#pragma unroll
            for (int j = 0; j < 4; ++j)
#pragma unroll
                for (int r = 0; r < 8; ++r) acc[t][j][r] = 0.f;

        for (int dc = 0; dc < 32; ++dc) {         // D reduction, 32 per step
            v16bf ah[4], al[4];
#pragma unroll
            for (int t = 0; t < 4; ++t) {
                ah[t] = Ah0[(size_t)t * 1024 + dc * 32];
                al[t] = Al0[(size_t)t * 1024 + dc * 32];
            }
            if (dc + 1 < 32)                      // global_prefetch next A
                __builtin_prefetch((const void*)(Ah0 + (dc + 1) * 32), 0, 1);
#pragma unroll
            for (int j = 0; j < 4; ++j) {
                size_t fb = ((size_t)(nbk * 4 + j) * 32 + dc) * 32;
                v16bf bh = Bh[fb];
                v16bf bl = Bl[fb];
#pragma unroll
                for (int t = 0; t < 4; ++t) {
                    acc[t][j] = __builtin_amdgcn_wmma_f32_16x16x32_bf16(
                        false, ah[t], false, bh, (short)0, acc[t][j], false, false);
                    acc[t][j] = __builtin_amdgcn_wmma_f32_16x16x32_bf16(
                        false, ah[t], false, bl, (short)0, acc[t][j], false, false);
                    acc[t][j] = __builtin_amdgcn_wmma_f32_16x16x32_bf16(
                        false, al[t], false, bh, (short)0, acc[t][j], false, false);
                }
            }
        }
        // dist[m,n] = ||E_n||^2 - 2*S[m,n]  (row-norm term is argmin-invariant)
#pragma unroll
        for (int j = 0; j < 4; ++j) {
            int n = (nbk * 4 + j) * 16 + (lane & 15);
            float en = e2[n];
#pragma unroll
            for (int t = 0; t < 4; ++t)
#pragma unroll
                for (int r = 0; r < 8; ++r) {
                    float dist = en - 2.0f * acc[t][j][r];
                    if (dist < minv[t][r]) { minv[t][r] = dist; mini[t][r] = n; }
                }
        }
    }

    // reduce across the 16 lanes that share each row, publish to LDS
#pragma unroll
    for (int t = 0; t < 4; ++t)
#pragma unroll
        for (int r = 0; r < 8; ++r) {
            float v = minv[t][r];
            int   i = mini[t][r];
#pragma unroll
            for (int off = 8; off > 0; off >>= 1) {
                float ov = __shfl_xor(v, off, 16);
                int   oi = __shfl_xor(i, off, 16);
                if (ov < v || (ov == v && oi < i)) { v = ov; i = oi; }
            }
            if ((lane & 15) == 0) {
                int rl = t * 16 + r + ((lane >> 4) << 3);   // local row 0..63
                s_val[mgroup][nq][rl] = v;
                s_idx[mgroup][nq][rl] = i;
            }
        }
    __syncthreads();

    // merge the 4 codebook-quarter partials per row (quarters have disjoint,
    // increasing n-ranges: strict < keeps the first/lowest index on ties)
    if (threadIdx.x < 128) {
        int mg = threadIdx.x >> 6;
        int rl = threadIdx.x & 63;
        float bv = s_val[mg][0][rl];
        int   bi = s_idx[mg][0][rl];
#pragma unroll
        for (int q = 1; q < 4; ++q) {
            float v = s_val[mg][q][rl];
            int   i = s_idx[mg][q][rl];
            if (v < bv) { bv = v; bi = i; }
        }
        idxout[blockIdx.x * 128 + mg * 64 + rl] = bi;
    }
}

// ---------------------------------------------------------------------------
// Kernel 5: gather exact f32 codebook columns, write q_out (bhwc order),
// discretised, and the fused loss  1.25 * mean((x - q)^2).
// ---------------------------------------------------------------------------
__global__ void vq_gather_out(const float* __restrict__ x,
                              const float* __restrict__ E,
                              const int* __restrict__ idx,
                              float* __restrict__ out,
                              float* __restrict__ disc,
                              float* __restrict__ lossp) {
    __shared__ float red[256];
    size_t i = (size_t)blockIdx.x * 256 + threadIdx.x;   // < 16777216
    int c = (int)(i & 255);
    int d = (int)((i >> 8) & 1023);
    int b = (int)(i >> 18);
    int m = b * CC + c;
    int q = idx[m];
    float e  = E[(size_t)d * KK + q];
    float xv = x[i];
    out[i] = e;                       // straight-through forward == q_bhwc
    float diff = xv - e;
    if (i < MM) disc[i] = (float)idx[i];

    red[threadIdx.x] = diff * diff;
    __syncthreads();
    for (int s = 128; s > 0; s >>= 1) {
        if (threadIdx.x < (unsigned)s) red[threadIdx.x] += red[threadIdx.x + s];
        __syncthreads();
    }
    if (threadIdx.x == 0)
        atomicAdd(lossp, red[0] * (1.25f / (float)NELEM));
}

// ---------------------------------------------------------------------------
extern "C" void kernel_launch(void* const* d_in, const int* in_sizes, int n_in,
                              void* d_out, int out_size, void* d_ws, size_t ws_size,
                              hipStream_t stream) {
    (void)in_sizes; (void)n_in; (void)out_size; (void)ws_size;
    const float* x = (const float*)d_in[0];           // [64,32,32,256] f32
    const float* E = (const float*)d_in[1];           // [1024,1024]    f32

    float* out   = (float*)d_out;                     // q_out: 16777216
    float* disc  = out + NELEM;                       // discretised: 16384
    float* lossp = disc + MM;                         // loss: 1

    // workspace carve-up (all 32B-aligned)
    char* ws = (char*)d_ws;
    float*          e2   = (float*)ws;                              // 4 KB
    int*            idxb = (int*)(ws + 4096);                       // 64 KB
    unsigned short* ebhi = (unsigned short*)(ws + 4096 + 65536);    // 2 MB
    unsigned short* eblo = ebhi + (size_t)KK * DD;                  // 2 MB
    unsigned short* xahi = eblo + (size_t)KK * DD;                  // 32 MB
    unsigned short* xalo = xahi + (size_t)MM * DD;                  // 32 MB

    vq_prep_e2    <<<4,     256, 0, stream>>>(E, e2, lossp);
    vq_prep_e     <<<128,   256, 0, stream>>>(E, ebhi, eblo);
    vq_prep_x     <<<2048,  256, 0, stream>>>(x, xahi, xalo);
    vq_gemm_argmin<<<128,   256, 0, stream>>>(xahi, xalo, ebhi, eblo, e2, idxb);
    vq_gather_out <<<NELEM / 256, 256, 0, stream>>>(x, E, idxb, out, disc, lossp);
}
